// MultiHeadAttentionWithKVCache_57939108823349
// MI455X (gfx1250) — compile-verified, matrix operand order checked by reading, not run on hardware
//
#include <hip/hip_runtime.h>
#include <hip/hip_bf16.h>

// ---------------------------------------------------------------------------
// Types / helpers
// ---------------------------------------------------------------------------
typedef __bf16 bf16_t;
typedef __attribute__((ext_vector_type(16))) __bf16 v16bf;
typedef __attribute__((ext_vector_type(8)))  float  v8f;
typedef __attribute__((ext_vector_type(4)))  unsigned v4u_t;
typedef __attribute__((ext_vector_type(8)))  unsigned v8u_t;

#define BATCH 2
#define SEQ   2048
#define HDIM  2048
#define NHEAD 16
#define DHEAD 128
#define MROWS (BATCH * SEQ)
#define ATTN_SCALE 0.08838834764831845f   // 1/sqrt(128)

// Flat pointer to LDS: low 32 bits of the generic address are the LDS offset.
#define LDS_OFF(p) ((unsigned)(unsigned long long)(p))

__device__ __forceinline__ bf16_t f2bf(float x) {
  union { float f; unsigned u; } a; a.f = x;
  unsigned r = a.u + 0x7FFFu + ((a.u >> 16) & 1u);   // round-to-nearest-even
  union { unsigned short s; bf16_t b; } o; o.s = (unsigned short)(r >> 16);
  return o.b;
}

union FragU { v16bf v; uint4 q[2]; };

// 16x32 bf16 WMMA operand fragment (documented 16-bit A layout; B uses the
// same per-lane pattern on the "row = column index" matrix).
__device__ __forceinline__ v16bf ld_frag(const bf16_t* p, int ld, int row,
                                         int k0, int lane) {
  const int base = (lane & 16) ? 8 : 0;
  const bf16_t* r = p + (size_t)row * ld + k0 + base;
  FragU f;
  f.q[0] = *(const uint4*)(r);
  f.q[1] = *(const uint4*)(r + 16);
  return f.v;
}

__device__ __forceinline__ v8f wmma_bf16(v16bf a, v16bf b, v8f c) {
  return __builtin_amdgcn_wmma_f32_16x16x32_bf16(
      false, a, false, b, (short)0, c, false, false);
}

__device__ __forceinline__ void zero8(v8f& v) {
#pragma unroll
  for (int i = 0; i < 8; ++i) v[i] = 0.0f;
}

// Per-lane async copy global -> LDS (ASYNCcnt tracked).
__device__ __forceinline__ void async_load_b128(unsigned lds, const void* g) {
  unsigned long long ga = (unsigned long long)g;
  asm volatile("global_load_async_to_lds_b128 %0, %1, off"
               :: "v"(lds), "v"(ga) : "memory");
}

// TDM: DMA a 2D tile (tile_d1 rows x tile_d0 bf16 elems, row stride
// stride_d0 elems) from global memory into LDS, padding each 16 stored
// DWORDs (one 32-elem row) with 4 DWORDs -> LDS row stride 40 elems.
__device__ __forceinline__ void tdm_load_tile_2d(unsigned lds_addr,
                                                 const void* gaddr,
                                                 unsigned tensor_d0,
                                                 unsigned tensor_d1,
                                                 unsigned tile_d0,
                                                 unsigned tile_d1,
                                                 unsigned stride_d0) {
  unsigned long long ga = (unsigned long long)gaddr;
  v4u_t g0;
  g0[0] = 1u;                                    // count=1, user descriptor
  g0[1] = lds_addr;                              // LDS byte address
  g0[2] = (unsigned)ga;                          // global addr [31:0]
  g0[3] = ((unsigned)(ga >> 32) & 0x1FFFFFFu) |  // global addr [56:32]
          (2u << 30);                            // type = image
  v8u_t g1;
  g1[0] = (1u << 16)      // data_size = 2 bytes
        | (1u << 20)      // pad_enable
        | (3u << 22)      // pad_interval: 16 DWORDs
        | (3u << 25);     // pad_amount: 4 DWORDs
  g1[1] = (tensor_d0 & 0xFFFFu) << 16;                       // dim0[15:0]
  g1[2] = (tensor_d0 >> 16) | ((tensor_d1 & 0xFFFFu) << 16); // dim0 hi,dim1 lo
  g1[3] = (tensor_d1 >> 16) | (tile_d0 << 16);               // dim1 hi,tile0
  g1[4] = tile_d1;                                           // tile1, tile2=0
  g1[5] = stride_d0;                                         // dim0 stride lo
  g1[6] = 0u;
  g1[7] = 0u;
  asm volatile("tensor_load_to_lds %0, %1" :: "s"(g0), "s"(g1) : "memory");
}

// ---------------------------------------------------------------------------
// fp32 -> bf16 conversion
// ---------------------------------------------------------------------------
__global__ void cvt_bf16_kernel(const float* __restrict__ in,
                                bf16_t* __restrict__ out, int n) {
  int i = (blockIdx.x * blockDim.x + threadIdx.x) * 4;
  if (i >= n) return;
  float4 v = *(const float4*)(in + i);
  union { ushort4 u; bf16_t b[4]; } o;
  o.b[0] = f2bf(v.x); o.b[1] = f2bf(v.y);
  o.b[2] = f2bf(v.z); o.b[3] = f2bf(v.w);
  *(ushort4*)(out + i) = o.u;
}

// ---------------------------------------------------------------------------
// RoPE cos/sin tables: [SEQ][DHEAD], value repeated in even/odd pairs
// ---------------------------------------------------------------------------
__global__ void rope_tables_kernel(float* __restrict__ cosT,
                                   float* __restrict__ sinT) {
  int s = blockIdx.x;
  int j = threadIdx.x;                       // 0..63
  float inv = expf(-(float)j * (logf(10000.0f) / 64.0f));
  float ang = (float)s * inv;
  float c = cosf(ang), sn = sinf(ang);
  int base = s * DHEAD + 2 * j;
  cosT[base] = c;  cosT[base + 1] = c;
  sinT[base] = sn; sinT[base + 1] = sn;
}

// ---------------------------------------------------------------------------
// Tiled WMMA GEMM:  out[m,n] = sum_k A[m,k] * W[n,k] + bias[n]
// Block 128x128, 8 waves (wave tile 32x64), K-step 32.
// A/W tiles staged by TDM (tensor_load_to_lds) with double-buffered LDS;
// wave 0 drives the DMA, everyone synchronizes via barrier after the
// leader's s_wait_tensorcnt.
// ---------------------------------------------------------------------------
#define LDS_STRIDE 40   // 32 + 4-DWORD TDM pad; 80B row stride (16B aligned)

__global__ __launch_bounds__(256)
void gemm_bf16_kernel(const bf16_t* __restrict__ A,
                      const bf16_t* __restrict__ W,
                      const float* __restrict__ bias,
                      bf16_t* __restrict__ outb, float* __restrict__ outf,
                      const float* __restrict__ cosT,
                      const float* __restrict__ sinT,
                      int M, int N, int K, int ropeFlag) {
  __shared__ bf16_t sA[2][128 * LDS_STRIDE];
  __shared__ bf16_t sB[2][128 * LDS_STRIDE];

  const int tid  = threadIdx.x;
  const int lane = tid & 31;
  const int wid  = tid >> 5;
  const int m0 = blockIdx.y * 128;
  const int n0 = blockIdx.x * 128;
  const int wm = (wid & 3) * 32;   // wave M offset inside block
  const int wn = (wid >> 2) * 64;  // wave N offset inside block
  const bool leader = (wid == 0);

  v8f acc[2][4];
#pragma unroll
  for (int i = 0; i < 2; ++i)
#pragma unroll
    for (int j = 0; j < 4; ++j) zero8(acc[i][j]);

  const bf16_t* Abase = A + (size_t)m0 * K;
  const bf16_t* Wbase = W + (size_t)n0 * K;
  const int nt = K / 32;

  if (leader) {   // prime tile 0
    tdm_load_tile_2d(LDS_OFF(&sA[0][0]), Abase, K, M, 32, 128, K);
    tdm_load_tile_2d(LDS_OFF(&sB[0][0]), Wbase, K, N, 32, 128, K);
  }

  for (int t = 0; t < nt; ++t) {
    if (leader) __builtin_amdgcn_s_wait_tensorcnt(0);
    __syncthreads();            // tile t visible; buffer (t+1)&1 free
    if (leader && (t + 1 < nt)) {
      const int k1 = (t + 1) * 32;
      tdm_load_tile_2d(LDS_OFF(&sA[(t + 1) & 1][0]), Abase + k1, K, M, 32, 128, K);
      tdm_load_tile_2d(LDS_OFF(&sB[(t + 1) & 1][0]), Wbase + k1, K, N, 32, 128, K);
    }
    const bf16_t* cA = sA[t & 1];
    const bf16_t* cB = sB[t & 1];

    v16bf af[2], bw[4];
#pragma unroll
    for (int mt = 0; mt < 2; ++mt)
      af[mt] = ld_frag(cA, LDS_STRIDE, wm + mt * 16 + (lane & 15), 0, lane);
#pragma unroll
    for (int nt2 = 0; nt2 < 4; ++nt2)
      bw[nt2] = ld_frag(cB, LDS_STRIDE, wn + nt2 * 16 + (lane & 15), 0, lane);
#pragma unroll
    for (int mt = 0; mt < 2; ++mt)
#pragma unroll
      for (int nt2 = 0; nt2 < 4; ++nt2)
        acc[mt][nt2] = wmma_bf16(af[mt], bw[nt2], acc[mt][nt2]);
  }

  // Epilogue: bias (+ optional fused RoPE) + store
  const int hl = lane >> 4;
#pragma unroll
  for (int mt = 0; mt < 2; ++mt) {
#pragma unroll
    for (int nt2 = 0; nt2 < 4; ++nt2) {
      v8f v = acc[mt][nt2];
      const int n = n0 + wn + nt2 * 16 + (lane & 15);
      const float b = bias[n];
#pragma unroll
      for (int r = 0; r < 8; ++r) {
        const int m = m0 + wm + mt * 16 + r + 8 * hl;
        float val = v[r] + b;
        if (ropeFlag) {
          float partner = __shfl_xor(val, 1, 32);  // even/odd RoPE pair
          int s  = m & (SEQ - 1);
          int di = n & (DHEAD - 1);
          float c  = cosT[s * DHEAD + di];
          float sn = sinT[s * DHEAD + di];
          val = (n & 1) ? (val * c + partner * sn) : (val * c - partner * sn);
        }
        if (outf) outf[(size_t)m * N + n] = val;
        else      outb[(size_t)m * N + n] = f2bf(val);
      }
    }
  }
}

// ---------------------------------------------------------------------------
// Flash attention (causal): 4 waves x 16 q-rows = 64 rows / workgroup,
// streaming 32-key tiles. K tile staged with async global->LDS copies;
// V tile transposed in flight (needs the VGPR round trip).
// ---------------------------------------------------------------------------
#define KSTR 136   // K tile row stride (128+8), 272B (16B aligned)
#define VSTR 40    // V^T / P row stride (32+8), 80B (16B aligned)

__global__ __launch_bounds__(128)
void flash_attn_kernel(const bf16_t* __restrict__ Q,
                       const bf16_t* __restrict__ Kv,
                       const bf16_t* __restrict__ V,
                       bf16_t* __restrict__ O) {
  __shared__ bf16_t sK[32 * KSTR];        // [key][dim]
  __shared__ bf16_t sVt[128 * VSTR];      // [dim][key]  (transposed V tile)
  __shared__ bf16_t sP[4][16 * VSTR];     // per-wave P tile [qrow][key]

  const int tid  = threadIdx.x;
  const int lane = tid & 31;
  const int wid  = tid >> 5;
  const int hl   = lane >> 4;
  const int bh = blockIdx.x;              // 0..B*NHEAD-1
  const int b  = bh / NHEAD;
  const int h  = bh % NHEAD;
  const int q0 = blockIdx.y * 64;         // workgroup q base
  const int qw = q0 + wid * 16;           // wave q base

  const bf16_t* Qb = Q + (size_t)(b * SEQ) * HDIM + h * DHEAD;
  const bf16_t* Kb = Kv + (size_t)(b * SEQ) * HDIM + h * DHEAD;
  const bf16_t* Vb = V + (size_t)(b * SEQ) * HDIM + h * DHEAD;

  // Preload the wave's 16x128 Q tile as four 16x32 A-fragments
  v16bf qf[4];
  const int qrow = qw + (lane & 15);
#pragma unroll
  for (int kk = 0; kk < 4; ++kk)
    qf[kk] = ld_frag(Qb, HDIM, qrow, kk * 32, lane);

  v8f accO[8];
#pragma unroll
  for (int i = 0; i < 8; ++i) zero8(accO[i]);
  float mrow[8], lrow[8];
#pragma unroll
  for (int r = 0; r < 8; ++r) { mrow[r] = -1e30f; lrow[r] = 0.0f; }

  const int nkt = q0 / 32 + 2;            // causal: key tiles covering q0+63
  for (int kc = 0; kc < nkt; ++kc) {
    const int kb = kc * 32;
    __syncthreads();                       // prior tile reads finished
    // Stage K tile (32x128) via async global->LDS (no VGPR round trip)
#pragma unroll
    for (int i = 0; i < 4; ++i) {
      int c = tid + i * 128, row = c >> 4, ch = (c & 15) * 8;
      async_load_b128(LDS_OFF(sK + row * KSTR + ch),
                      Kb + (size_t)(kb + row) * HDIM + ch);
    }
    // Stage V tile transposed: sVt[dim][key]
#pragma unroll
    for (int i = 0; i < 4; ++i) {
      int c = tid + i * 128, row = c >> 4, ch = (c & 15) * 8;
      uint4 u = *(const uint4*)(Vb + (size_t)(kb + row) * HDIM + ch);
      const bf16_t* e = (const bf16_t*)&u;
#pragma unroll
      for (int j = 0; j < 8; ++j) sVt[(ch + j) * VSTR + row] = e[j];
    }
    asm volatile("s_wait_asynccnt 0x0" ::: "memory");
    __syncthreads();

    // S = Q K^T for 32 keys (two 16x16 tiles), K-dim = 128 (4 WMMA steps)
    v8f s0, s1; zero8(s0); zero8(s1);
#pragma unroll
    for (int kk = 0; kk < 4; ++kk) {
      v16bf bk0 = ld_frag(sK, KSTR, (lane & 15),      kk * 32, lane);
      v16bf bk1 = ld_frag(sK, KSTR, 16 + (lane & 15), kk * 32, lane);
      s0 = wmma_bf16(qf[kk], bk0, s0);
      s1 = wmma_bf16(qf[kk], bk1, s1);
    }

    // Scale, causal mask, online softmax (row stats per 16-lane half)
    float corr[8];
#pragma unroll
    for (int r = 0; r < 8; ++r) {
      const int q    = qw + r + 8 * hl;
      const int key0 = kb + (lane & 15);
      float a0 = s0[r] * ATTN_SCALE; if (key0 > q)      a0 = -1e30f;
      float a1 = s1[r] * ATTN_SCALE; if (key0 + 16 > q) a1 = -1e30f;
      float mx = fmaxf(a0, a1);
#pragma unroll
      for (int off = 8; off >= 1; off >>= 1)
        mx = fmaxf(mx, __shfl_xor(mx, off, 32));
      const float mnew = fmaxf(mrow[r], mx);
      corr[r] = __expf(mrow[r] - mnew);
      mrow[r] = mnew;
      a0 = __expf(a0 - mnew);
      a1 = __expf(a1 - mnew);
      s0[r] = a0; s1[r] = a1;
      float rs = a0 + a1;
#pragma unroll
      for (int off = 8; off >= 1; off >>= 1)
        rs += __shfl_xor(rs, off, 32);
      lrow[r] = lrow[r] * corr[r] + rs;
    }
#pragma unroll
    for (int dt = 0; dt < 8; ++dt)
#pragma unroll
      for (int r = 0; r < 8; ++r) accO[dt][r] *= corr[r];

    // Re-layout P (C-layout -> A-layout) through per-wave LDS
#pragma unroll
    for (int r = 0; r < 8; ++r) {
      const int ml = r + 8 * hl;
      sP[wid][ml * VSTR + (lane & 15)]      = f2bf(s0[r]);
      sP[wid][ml * VSTR + 16 + (lane & 15)] = f2bf(s1[r]);
    }
    asm volatile("s_wait_dscnt 0x0" ::: "memory");  // P stores -> P loads

    // O += P V  (k-dim = 32 keys, one WMMA step per d-tile)
    v16bf ap = ld_frag(sP[wid], VSTR, (lane & 15), 0, lane);
#pragma unroll
    for (int dt = 0; dt < 8; ++dt) {
      v16bf bv = ld_frag(sVt, VSTR, dt * 16 + (lane & 15), 0, lane);
      accO[dt] = wmma_bf16(ap, bv, accO[dt]);
    }
  }

  // Normalize and store attention output (bf16, [B*S, H] layout)
  bf16_t* Ob = O + (size_t)(b * SEQ) * HDIM + h * DHEAD;
#pragma unroll
  for (int r = 0; r < 8; ++r) {
    const float inv = 1.0f / lrow[r];
    const int q = qw + r + 8 * hl;
#pragma unroll
    for (int dt = 0; dt < 8; ++dt) {
      const int n = dt * 16 + (lane & 15);
      Ob[(size_t)q * HDIM + n] = f2bf(accO[dt][r] * inv);
    }
  }
}

// ---------------------------------------------------------------------------
// Launch
// ---------------------------------------------------------------------------
extern "C" void kernel_launch(void* const* d_in, const int* in_sizes, int n_in,
                              void* d_out, int out_size, void* d_ws,
                              size_t ws_size, hipStream_t stream) {
  (void)in_sizes; (void)n_in; (void)out_size; (void)ws_size;
  const float* x  = (const float*)d_in[0];
  const float* Wq = (const float*)d_in[1];
  const float* bq = (const float*)d_in[2];
  const float* Wk = (const float*)d_in[3];
  const float* bk = (const float*)d_in[4];
  const float* Wv = (const float*)d_in[5];
  const float* bv = (const float*)d_in[6];
  const float* Wo = (const float*)d_in[7];
  const float* bo = (const float*)d_in[8];
  float* out = (float*)d_out;

  size_t off = 0;
  auto carve = [&](size_t bytes) -> void* {
    void* p = (char*)d_ws + off;
    off += (bytes + 255) & ~(size_t)255;
    return p;
  };
  const size_t MK = (size_t)MROWS * HDIM;   // 4096*2048
  const size_t WW = (size_t)HDIM * HDIM;    // 2048*2048
  bf16_t* xb  = (bf16_t*)carve(MK * 2);
  bf16_t* wqb = (bf16_t*)carve(WW * 2);
  bf16_t* wkb = (bf16_t*)carve(WW * 2);
  bf16_t* wvb = (bf16_t*)carve(WW * 2);
  bf16_t* wob = (bf16_t*)carve(WW * 2);
  bf16_t* qb  = (bf16_t*)carve(MK * 2);
  bf16_t* kb  = (bf16_t*)carve(MK * 2);
  bf16_t* vb  = (bf16_t*)carve(MK * 2);
  bf16_t* ab  = (bf16_t*)carve(MK * 2);
  float* cosT = (float*)carve((size_t)SEQ * DHEAD * 4);
  float* sinT = (float*)carve((size_t)SEQ * DHEAD * 4);

  // 1) fp32 -> bf16
  cvt_bf16_kernel<<<(int)(MK / 4 / 256), 256, 0, stream>>>(x, xb, (int)MK);
  cvt_bf16_kernel<<<(int)(WW / 4 / 256), 256, 0, stream>>>(Wq, wqb, (int)WW);
  cvt_bf16_kernel<<<(int)(WW / 4 / 256), 256, 0, stream>>>(Wk, wkb, (int)WW);
  cvt_bf16_kernel<<<(int)(WW / 4 / 256), 256, 0, stream>>>(Wv, wvb, (int)WW);
  cvt_bf16_kernel<<<(int)(WW / 4 / 256), 256, 0, stream>>>(Wo, wob, (int)WW);

  // 2) RoPE tables
  rope_tables_kernel<<<SEQ, DHEAD / 2, 0, stream>>>(cosT, sinT);

  // 3) Q/K/V projections (RoPE fused into Q and K epilogues)
  dim3 ggrid(HDIM / 128, MROWS / 128);
  gemm_bf16_kernel<<<ggrid, 256, 0, stream>>>(xb, wqb, bq, qb, nullptr,
                                              cosT, sinT, MROWS, HDIM, HDIM, 1);
  gemm_bf16_kernel<<<ggrid, 256, 0, stream>>>(xb, wkb, bk, kb, nullptr,
                                              cosT, sinT, MROWS, HDIM, HDIM, 1);
  gemm_bf16_kernel<<<ggrid, 256, 0, stream>>>(xb, wvb, bv, vb, nullptr,
                                              cosT, sinT, MROWS, HDIM, HDIM, 0);

  // 4) Causal flash attention
  dim3 fgrid(BATCH * NHEAD, SEQ / 64);
  flash_attn_kernel<<<fgrid, 128, 0, stream>>>(qb, kb, vb, ab);

  // 5) Output projection -> fp32 d_out
  gemm_bf16_kernel<<<ggrid, 256, 0, stream>>>(ab, wob, bo, nullptr, out,
                                              cosT, sinT, MROWS, HDIM, HDIM, 0);
}
